// ViT_Regression_24455543783438
// MI455X (gfx1250) — matrix operand order repaired and checked by reading, not compiled, not run on hardware
//
#include <hip/hip_runtime.h>
#include <hip/hip_bf16.h>

#define B_    16
#define SEQ_  513
#define D_    768
#define NH_   12
#define DH_   64
#define L_    12
#define FF_   3072
#define NPATCH_ 512
#define NSP_  128
#define SEQP_ 544     // padded seq (17*32) for attention tiles

typedef __attribute__((ext_vector_type(16))) __bf16 v16bf;
typedef __attribute__((ext_vector_type(8)))  float  v8f;

union FragB16 { v16bf v; unsigned u[8]; };
union FragF32 { v8f v; float f[8]; };

__device__ inline v8f wmma_bf16(v16bf a, v16bf b, v8f c) {
  return __builtin_amdgcn_wmma_f32_16x16x32_bf16(false, a, false, b, (short)0, c, false, false);
}

// xor-butterfly within each 16-lane row on the VALU crossbar (v_permlane16_b32)
// instead of LDS ds_bpermute. Selector nibbles encode lane i -> lane i^d.
#if __has_builtin(__builtin_amdgcn_permlane16)
__device__ inline float pl16_xor(float x, unsigned lo, unsigned hi) {
  unsigned xi = __builtin_bit_cast(unsigned, x);
  unsigned r = __builtin_amdgcn_permlane16(xi, xi, lo, hi, false, false);
  return __builtin_bit_cast(float, r);
}
__device__ inline float rowmax16(float x) {
  x = fmaxf(x, pl16_xor(x, 0x67452301u, 0xEFCDAB89u));  // xor 1
  x = fmaxf(x, pl16_xor(x, 0x54761032u, 0xDCFE98BAu));  // xor 2
  x = fmaxf(x, pl16_xor(x, 0x32107654u, 0xBA98FEDCu));  // xor 4
  x = fmaxf(x, pl16_xor(x, 0xFEDCBA98u, 0x76543210u));  // xor 8
  return x;
}
#else
__device__ inline float rowmax16(float x) {
  for (int d = 1; d < 16; d <<= 1) x = fmaxf(x, __shfl_xor(x, d, 32));
  return x;
}
#endif

// ---------------- fp32 -> bf16 conversion ----------------
__global__ __launch_bounds__(256) void cvt_bf16(const float* __restrict__ in,
                                                __bf16* __restrict__ out, int n) {
  int i = blockIdx.x * 256 + threadIdx.x;
  if (i < n) out[i] = (__bf16)in[i];
}

// ---------------- general WMMA GEMM ----------------
// C[M,N] = A[M,K](bf16,row) @ B[K,N](bf16,row)  (+bias)(gelu)(+resid) -> f32 or bf16
// 256 threads = 8 waves (4m x 2n), block tile 128x128, wave tile 32x64, K-step 32
#define BM 128
#define BN 128
#define BK 32
template<bool BIAS, bool GELU, bool RESID, bool OUTBF>
__global__ __launch_bounds__(256) void gemm_wmma(
    const __bf16* __restrict__ A, const __bf16* __restrict__ Bm,
    const float* __restrict__ bias, float* Cf, __bf16* Cb, const float* Rz,
    int M, int N, int K)
{
  __shared__ __bf16 As[BM][BK + 2];
  __shared__ __bf16 Bs[BN][BK + 2];
  const int tid = threadIdx.x;
  const int bm0 = blockIdx.y * BM;
  const int bn0 = blockIdx.x * BN;
  const int wave = tid >> 5, lane = tid & 31;
  const int wm = wave >> 1, wn = wave & 1;     // 4 x 2 waves
  const int g = lane >> 4, ln = lane & 15;

  FragF32 acc[2][4];
#pragma unroll
  for (int mi = 0; mi < 2; ++mi)
#pragma unroll
    for (int ni = 0; ni < 4; ++ni)
#pragma unroll
      for (int j = 0; j < 8; ++j) acc[mi][ni].f[j] = 0.f;

  for (int k0 = 0; k0 < K; k0 += BK) {
    __syncthreads();
    // ---- A tile 128x32: two 16B chunks per thread ----
#pragma unroll
    for (int aa = 0; aa < 2; ++aa) {
      int r = (tid >> 2) + aa * 64;
      int c = (tid & 3) * 8;
      int gr = bm0 + r;
      unsigned* dst = (unsigned*)&As[r][c];
      if (gr < M) {
        uint4 d = *(const uint4*)(A + (size_t)gr * K + k0 + c);
        dst[0] = d.x; dst[1] = d.y; dst[2] = d.z; dst[3] = d.w;
      } else {
        dst[0] = dst[1] = dst[2] = dst[3] = 0u;
      }
    }
    // ---- B tile 32x128 transposed into Bs[n][k]: 2 chunks per thread ----
#pragma unroll
    for (int cc = 0; cc < 2; ++cc) {
      int c = tid + cc * 256;          // 0..511
      int kr = c >> 4;                 // 0..31
      int nc = (c & 15) * 8;           // 0..120
      union { uint4 q; __bf16 h[8]; } tmp;
      tmp.q = *(const uint4*)(Bm + (size_t)(k0 + kr) * N + bn0 + nc);
#pragma unroll
      for (int u = 0; u < 8; ++u) Bs[nc + u][kr] = tmp.h[u];
    }
    // prefetch next K-step tiles into cache (global_prefetch_b8, speculative)
    if (k0 + BK < K) {
      int gr = bm0 + (tid >> 2);
      if (gr < M) __builtin_prefetch(A + (size_t)gr * K + k0 + BK, 0, 3);
      __builtin_prefetch(Bm + (size_t)(k0 + BK + (tid >> 4)) * N + bn0 + (tid & 15) * 8, 0, 3);
    }
    __syncthreads();

    FragB16 af[2], bf[4];
#pragma unroll
    for (int mi = 0; mi < 2; ++mi) {
      int m = wm * 32 + mi * 16 + ln;
#pragma unroll
      for (int p = 0; p < 8; ++p) {
        int k = 16 * (p >> 2) + 8 * g + (p & 3) * 2;
        af[mi].u[p] = *(const unsigned*)&As[m][k];
      }
    }
#pragma unroll
    for (int ni = 0; ni < 4; ++ni) {
      int n = wn * 64 + ni * 16 + ln;
#pragma unroll
      for (int p = 0; p < 8; ++p) {
        int k = 16 * (p >> 2) + 8 * g + (p & 3) * 2;
        bf[ni].u[p] = *(const unsigned*)&Bs[n][k];
      }
    }
#pragma unroll
    for (int mi = 0; mi < 2; ++mi)
#pragma unroll
      for (int ni = 0; ni < 4; ++ni)
        acc[mi][ni].v = wmma_bf16(af[mi].v, bf[ni].v, acc[mi][ni].v);
  }

  // ---- epilogue ----
#pragma unroll
  for (int mi = 0; mi < 2; ++mi)
#pragma unroll
    for (int ni = 0; ni < 4; ++ni) {
      int col = bn0 + wn * 64 + ni * 16 + ln;
      float bv = BIAS ? bias[col] : 0.f;
#pragma unroll
      for (int j = 0; j < 8; ++j) {
        int row = bm0 + wm * 32 + mi * 16 + 8 * g + j;
        if (row < M) {
          float v = acc[mi][ni].f[j] + bv;
          if (GELU) v = 0.5f * v * (1.f + erff(v * 0.70710678f));
          size_t off = (size_t)row * N + col;
          if (RESID) v += Rz[off];
          if (OUTBF) Cb[off] = (__bf16)v; else Cf[off] = v;
        }
      }
    }
}

// ---------------- assemble z = pos + [strain|s] + [cls|e] ----------------
__global__ __launch_bounds__(256) void build_z(
    const float* __restrict__ e, const float* __restrict__ s_emb,
    const float* __restrict__ cls, const float* __restrict__ pos,
    const float* __restrict__ strain, float* __restrict__ z)
{
  int idx = blockIdx.x * 256 + threadIdx.x;
  if (idx >= B_ * SEQ_ * D_) return;
  int d = idx % D_;
  int t = (idx / D_) % SEQ_;
  int b = idx / (D_ * SEQ_);
  float v = pos[t * D_ + d];
  v += (t < 385) ? strain[t * D_ + d]
                 : s_emb[((size_t)b * NSP_ + (t - 385)) * D_ + d];
  v += (t == 0) ? cls[d]
                : e[((size_t)b * NPATCH_ + (t - 1)) * D_ + d];
  z[idx] = v;
}

// ---------------- layernorm (f32 in -> bf16 out), one wave per token ----------------
__global__ __launch_bounds__(256) void layernorm_bf16(
    const float* __restrict__ z, const float* __restrict__ sc,
    const float* __restrict__ bi, __bf16* __restrict__ out, int ntok)
{
  int token = blockIdx.x * 8 + (threadIdx.x >> 5);
  if (token >= ntok) return;
  int lane = threadIdx.x & 31;
  const float* row = z + (size_t)token * D_;
  float v[24], s = 0.f, s2 = 0.f;
#pragma unroll
  for (int i = 0; i < 24; ++i) {
    v[i] = row[lane + i * 32];
    s += v[i]; s2 += v[i] * v[i];
  }
#pragma unroll
  for (int d = 1; d < 32; d <<= 1) {
    s += __shfl_xor(s, d, 32);
    s2 += __shfl_xor(s2, d, 32);
  }
  float mean = s * (1.f / D_);
  float var = s2 * (1.f / D_) - mean * mean;
  float rstd = rsqrtf(var + 1e-5f);
  __bf16* orow = out + (size_t)token * D_;
#pragma unroll
  for (int i = 0; i < 24; ++i) {
    int d = lane + i * 32;
    orow[d] = (__bf16)((v[i] - mean) * rstd * sc[d] + bi[d]);
  }
}

// ---------------- split qkv (torch (h d 3) layout) into Q,K (s-major) and V (d-major) ----------------
__global__ __launch_bounds__(256) void split_qkv(
    const __bf16* __restrict__ qkv, __bf16* __restrict__ Qb,
    __bf16* __restrict__ Kb, __bf16* __restrict__ Vt)
{
  int idx = blockIdx.x * 256 + threadIdx.x;
  if (idx >= B_ * NH_ * SEQP_ * DH_) return;
  int d = idx % DH_;
  int s = (idx / DH_) % SEQP_;
  int h = (idx / (DH_ * SEQP_)) % NH_;
  int b = idx / (DH_ * SEQP_ * NH_);
  size_t qo = ((size_t)(b * NH_ + h) * SEQP_ + s) * DH_ + d;
  size_t vo = ((size_t)(b * NH_ + h) * DH_ + d) * SEQP_ + s;
  if (s < SEQ_) {
    const __bf16* base = qkv + ((size_t)b * SEQ_ + s) * (3 * D_) + h * (3 * DH_) + d * 3;
    Qb[qo] = (__bf16)((float)base[0] * 0.125f);   // fold 1/sqrt(DH)
    Kb[qo] = base[1];
    Vt[vo] = base[2];
  } else {
    Qb[qo] = (__bf16)0.f; Kb[qo] = (__bf16)0.f; Vt[vo] = (__bf16)0.f;
  }
}

// ---------------- flash attention: wave per 16-query tile, K-tile=32 ----------------
// Row-sums come from an extra WMMA against an all-ones B fragment (l = P @ 1);
// row-max butterfly runs on v_permlane16 (VALU) instead of LDS bpermute.
__global__ __launch_bounds__(128) void attn_flash(
    const __bf16* __restrict__ Qb, const __bf16* __restrict__ Kb,
    const __bf16* __restrict__ Vt, float* z)
{
  __shared__ __bf16 Pbuf[4][16][34];
  const int w = threadIdx.x >> 5, lane = threadIdx.x & 31;
  const int g = lane >> 4, ln = lane & 15;
  const int bh = blockIdx.x;
  const int b = bh / NH_, h = bh % NH_;
  const int qt = blockIdx.y * 4 + w;
  if (qt >= 33) return;   // no block barriers below -> safe

  const __bf16* Qp = Qb + ((size_t)bh * SEQP_ + qt * 16) * DH_;
  const __bf16* Kp = Kb + (size_t)bh * SEQP_ * DH_;
  const __bf16* Vp = Vt + (size_t)bh * DH_ * SEQP_;

  // Q fragments (two d-halves)
  FragB16 qa[2];
#pragma unroll
  for (int dd = 0; dd < 2; ++dd)
#pragma unroll
    for (int p = 0; p < 8; ++p) {
      int k = dd * 32 + 16 * (p >> 2) + 8 * g + (p & 3) * 2;
      qa[dd].u[p] = *(const unsigned*)(Qp + ln * DH_ + k);
    }

  FragB16 ones;
#pragma unroll
  for (int p = 0; p < 8; ++p) ones.u[p] = 0x3F803F80u;  // bf16 1.0 pair

  float mrow[8];
  FragF32 o[5];               // o[0..3] = output d-chunks, o[4] = row-sum accumulator
#pragma unroll
  for (int j = 0; j < 8; ++j) mrow[j] = -INFINITY;
#pragma unroll
  for (int dc = 0; dc < 5; ++dc)
#pragma unroll
    for (int j = 0; j < 8; ++j) o[dc].f[j] = 0.f;

  for (int kt = 0; kt < 17; ++kt) {
    // S = Q @ K^T for 32 keys (two 16-col C tiles)
    FragB16 kf[2][2];
#pragma unroll
    for (int ni = 0; ni < 2; ++ni) {
      int key = kt * 32 + ni * 16 + ln;
      const __bf16* kr = Kp + (size_t)key * DH_;
#pragma unroll
      for (int dd = 0; dd < 2; ++dd)
#pragma unroll
        for (int p = 0; p < 8; ++p)
          kf[ni][dd].u[p] = *(const unsigned*)(kr + dd * 32 + 16 * (p >> 2) + 8 * g + (p & 3) * 2);
    }
    FragF32 s0, s1;
    { v8f t = {}; t = wmma_bf16(qa[0].v, kf[0][0].v, t); t = wmma_bf16(qa[1].v, kf[0][1].v, t); s0.v = t; }
    { v8f t = {}; t = wmma_bf16(qa[0].v, kf[1][0].v, t); t = wmma_bf16(qa[1].v, kf[1][1].v, t); s1.v = t; }

    int key0 = kt * 32 + ln, key1 = key0 + 16;
    if (key0 >= SEQ_) {
#pragma unroll
      for (int j = 0; j < 8; ++j) s0.f[j] = -1e30f;
    }
    if (key1 >= SEQ_) {
#pragma unroll
      for (int j = 0; j < 8; ++j) s1.f[j] = -1e30f;
    }

    float pr0[8], pr1[8];
#pragma unroll
    for (int j = 0; j < 8; ++j) {
      float mx = rowmax16(fmaxf(s0.f[j], s1.f[j]));
      float mn = fmaxf(mrow[j], mx);
      float scale = __expf(mrow[j] - mn);
      pr0[j] = __expf(s0.f[j] - mn);
      pr1[j] = __expf(s1.f[j] - mn);
      mrow[j] = mn;
#pragma unroll
      for (int dc = 0; dc < 5; ++dc) o[dc].f[j] *= scale;
    }

    // C-layout -> A-layout for P via per-wave LDS
#pragma unroll
    for (int j = 0; j < 8; ++j) {
      Pbuf[w][8 * g + j][ln]      = (__bf16)pr0[j];
      Pbuf[w][8 * g + j][16 + ln] = (__bf16)pr1[j];
    }
    asm volatile("s_wait_dscnt 0" ::: "memory");
    FragB16 pa;
#pragma unroll
    for (int p = 0; p < 8; ++p) {
      int k = 16 * (p >> 2) + 8 * g + (p & 3) * 2;
      pa.u[p] = *(const unsigned*)&Pbuf[w][ln][k];
    }
    // O += P @ V  (V stored d-major so B frags are contiguous dwords)
#pragma unroll
    for (int dc = 0; dc < 4; ++dc) {
      FragB16 vf;
      const __bf16* vr = Vp + (size_t)(dc * 16 + ln) * SEQP_ + kt * 32;
#pragma unroll
      for (int p = 0; p < 8; ++p)
        vf.u[p] = *(const unsigned*)(vr + 16 * (p >> 2) + 8 * g + (p & 3) * 2);
      o[dc].v = wmma_bf16(pa.v, vf.v, o[dc].v);
    }
    // l += P @ ones  (row-sum on the matrix core)
    o[4].v = wmma_bf16(pa.v, ones.v, o[4].v);
  }

  // residual add into z
#pragma unroll
  for (int j = 0; j < 8; ++j) {
    int s = qt * 16 + 8 * g + j;
    if (s < SEQ_) {
      float inv = 1.f / o[4].f[j];
      size_t base = ((size_t)b * SEQ_ + s) * D_ + h * DH_;
#pragma unroll
      for (int dc = 0; dc < 4; ++dc) {
        size_t off = base + dc * 16 + ln;
        z[off] += o[dc].f[j] * inv;
      }
    }
  }
}

// ---------------- mean pool over tokens ----------------
__global__ __launch_bounds__(256) void pool_mean(const float* __restrict__ z,
                                                 float* __restrict__ y) {
  int idx = blockIdx.x * 256 + threadIdx.x;
  if (idx >= B_ * D_) return;
  int b = idx / D_, d = idx % D_;
  float s = 0.f;
  for (int t = 0; t < SEQ_; ++t) s += z[((size_t)b * SEQ_ + t) * D_ + d];
  y[idx] = s * (1.f / SEQ_);
}

// ---------------- head: LN -> 768x256 -> 256x64 -> 64x1 ----------------
__global__ __launch_bounds__(256) void head_mlp(
    const float* __restrict__ y, const float* __restrict__ hn_s,
    const float* __restrict__ hn_b, const float* __restrict__ h1w,
    const float* __restrict__ h1b, const float* __restrict__ h2w,
    const float* __restrict__ h2b, const float* __restrict__ h3w,
    const float* __restrict__ h3b, float* __restrict__ out)
{
  __shared__ float yb[768];
  __shared__ float t1[256];
  __shared__ float t2[64];
  __shared__ float red[256];
  int b = blockIdx.x, t = threadIdx.x;
  float s = 0.f, s2 = 0.f;
  for (int i = t; i < 768; i += 256) {
    float v = y[(size_t)b * 768 + i];
    yb[i] = v; s += v; s2 += v * v;
  }
  red[t] = s; __syncthreads();
  for (int st = 128; st > 0; st >>= 1) { if (t < st) red[t] += red[t + st]; __syncthreads(); }
  float mean = red[0] * (1.f / 768.f);
  __syncthreads();
  red[t] = s2; __syncthreads();
  for (int st = 128; st > 0; st >>= 1) { if (t < st) red[t] += red[t + st]; __syncthreads(); }
  float var = red[0] * (1.f / 768.f) - mean * mean;
  float rstd = rsqrtf(var + 1e-5f);
  __syncthreads();
  for (int i = t; i < 768; i += 256) yb[i] = (yb[i] - mean) * rstd * hn_s[i] + hn_b[i];
  __syncthreads();
  {
    float acc = h1b[t];
    for (int d = 0; d < 768; ++d) acc += yb[d] * h1w[(size_t)d * 256 + t];
    t1[t] = acc;
  }
  __syncthreads();
  if (t < 64) {
    float acc = h2b[t];
    for (int i = 0; i < 256; ++i) acc += t1[i] * h2w[(size_t)i * 64 + t];
    t2[t] = acc;
  }
  __syncthreads();
  if (t == 0) {
    float acc = h3b[0];
    for (int i = 0; i < 64; ++i) acc += t2[i] * h3w[i];
    out[b] = acc;
  }
}

// ---------------- workspace layout (bytes, all 256-aligned) ----------------
static constexpr size_t OFF_PWB  = 0;                         // 256*768 bf16
static constexpr size_t OFF_IN1B = OFF_PWB  + 393216;         // 16*512*256 bf16
static constexpr size_t OFF_IN2B = OFF_IN1B + 4194304;        // 16*128*256 bf16
static constexpr size_t OFF_Z    = OFF_IN2B + 1048576;        // 16*513*768 f32
static constexpr size_t OFF_XN   = OFF_Z    + 25214976;       // 16*513*768 bf16
static constexpr size_t OFF_WB   = OFF_XN   + 12607488;       // 768*3072 bf16
static constexpr size_t OFF_RB   = OFF_WB   + 4718592;        // union{e+s , Q/K/Vt} = 40108032
static constexpr size_t OFF_RC   = OFF_RB   + 40108032;       // union{qkv_bf , ff_bf} = 50429952
static constexpr size_t OFF_Y    = OFF_RC   + 50429952;       // 16*768 f32

extern "C" void kernel_launch(void* const* d_in, const int* in_sizes, int n_in,
                              void* d_out, int out_size, void* d_ws, size_t ws_size,
                              hipStream_t stream) {
  const float* input1  = (const float*)d_in[0];
  const float* input2  = (const float*)d_in[1];
  const float* proj_w  = (const float*)d_in[2];
  const float* proj_b  = (const float*)d_in[3];
  const float* cls_tok = (const float*)d_in[4];
  const float* pos_emb = (const float*)d_in[5];
  const float* strain  = (const float*)d_in[6];
  const float* ln1_s   = (const float*)d_in[7];
  const float* ln1_b   = (const float*)d_in[8];
  const float* wqkv    = (const float*)d_in[9];
  const float* ln2_s   = (const float*)d_in[10];
  const float* ln2_b   = (const float*)d_in[11];
  const float* fc1_w   = (const float*)d_in[12];
  const float* fc1_b   = (const float*)d_in[13];
  const float* fc2_w   = (const float*)d_in[14];
  const float* fc2_b   = (const float*)d_in[15];
  const float* hn_s    = (const float*)d_in[16];
  const float* hn_b    = (const float*)d_in[17];
  const float* h1_w    = (const float*)d_in[18];
  const float* h1_b    = (const float*)d_in[19];
  const float* h2_w    = (const float*)d_in[20];
  const float* h2_b    = (const float*)d_in[21];
  const float* h3_w    = (const float*)d_in[22];
  const float* h3_b    = (const float*)d_in[23];
  float* out = (float*)d_out;

  char* ws = (char*)d_ws;
  __bf16* pwb  = (__bf16*)(ws + OFF_PWB);
  __bf16* in1b = (__bf16*)(ws + OFF_IN1B);
  __bf16* in2b = (__bf16*)(ws + OFF_IN2B);
  float*  z    = (float*)(ws + OFF_Z);
  __bf16* xn   = (__bf16*)(ws + OFF_XN);
  __bf16* wb   = (__bf16*)(ws + OFF_WB);
  float*  e    = (float*)(ws + OFF_RB);
  float*  semb = (float*)(ws + OFF_RB + 25165824);
  __bf16* Qb   = (__bf16*)(ws + OFF_RB);
  __bf16* Kb   = (__bf16*)(ws + OFF_RB + 13369344);
  __bf16* Vt   = (__bf16*)(ws + OFF_RB + 26738688);
  __bf16* qkvb = (__bf16*)(ws + OFF_RC);
  __bf16* ffb  = (__bf16*)(ws + OFF_RC);
  float*  y    = (float*)(ws + OFF_Y);

  auto cdiv = [](int a, int b) { return (a + b - 1) / b; };

  // ---- patch embedding ----
  cvt_bf16<<<cdiv(256 * 768, 256), 256, 0, stream>>>(proj_w, pwb, 256 * 768);
  cvt_bf16<<<cdiv(B_ * NPATCH_ * 256, 256), 256, 0, stream>>>(input1, in1b, B_ * NPATCH_ * 256);
  cvt_bf16<<<cdiv(B_ * NSP_ * 256, 256), 256, 0, stream>>>(input2, in2b, B_ * NSP_ * 256);

  {
    int M = B_ * NPATCH_, N = D_, K = 256;
    dim3 grid(N / BN, cdiv(M, BM));
    gemm_wmma<true, false, false, false><<<grid, 256, 0, stream>>>(
        in1b, pwb, proj_b, e, nullptr, nullptr, M, N, K);
  }
  {
    int M = B_ * NSP_, N = D_, K = 256;
    dim3 grid(N / BN, cdiv(M, BM));
    gemm_wmma<true, false, false, false><<<grid, 256, 0, stream>>>(
        in2b, pwb, proj_b, semb, nullptr, nullptr, M, N, K);
  }
  build_z<<<cdiv(B_ * SEQ_ * D_, 256), 256, 0, stream>>>(e, semb, cls_tok, pos_emb, strain, z);

  const int NTOK = B_ * SEQ_;   // 8208
  // ---- transformer layers ----
  for (int l = 0; l < L_; ++l) {
    // attention branch
    layernorm_bf16<<<cdiv(NTOK, 8), 256, 0, stream>>>(z, ln1_s + l * D_, ln1_b + l * D_, xn, NTOK);
    cvt_bf16<<<cdiv(D_ * 3 * D_, 256), 256, 0, stream>>>(wqkv + (size_t)l * D_ * 3 * D_, wb, D_ * 3 * D_);
    {
      int M = NTOK, N = 3 * D_, K = D_;
      dim3 grid(N / BN, cdiv(M, BM));
      gemm_wmma<false, false, false, true><<<grid, 256, 0, stream>>>(
          xn, wb, nullptr, nullptr, qkvb, nullptr, M, N, K);
    }
    split_qkv<<<cdiv(B_ * NH_ * SEQP_ * DH_, 256), 256, 0, stream>>>(qkvb, Qb, Kb, Vt);
    {
      dim3 grid(B_ * NH_, 9);   // 9*4 waves >= 33 q-tiles
      attn_flash<<<grid, 128, 0, stream>>>(Qb, Kb, Vt, z);
    }
    // MLP branch
    layernorm_bf16<<<cdiv(NTOK, 8), 256, 0, stream>>>(z, ln2_s + l * D_, ln2_b + l * D_, xn, NTOK);
    cvt_bf16<<<cdiv(D_ * FF_, 256), 256, 0, stream>>>(fc1_w + (size_t)l * D_ * FF_, wb, D_ * FF_);
    {
      int M = NTOK, N = FF_, K = D_;
      dim3 grid(N / BN, cdiv(M, BM));
      gemm_wmma<true, true, false, true><<<grid, 256, 0, stream>>>(
          xn, wb, fc1_b + l * FF_, nullptr, ffb, nullptr, M, N, K);
    }
    cvt_bf16<<<cdiv(FF_ * D_, 256), 256, 0, stream>>>(fc2_w + (size_t)l * FF_ * D_, wb, FF_ * D_);
    {
      int M = NTOK, N = D_, K = FF_;
      dim3 grid(N / BN, cdiv(M, BM));
      gemm_wmma<true, false, true, false><<<grid, 256, 0, stream>>>(
          ffb, wb, fc2_b + l * D_, z, nullptr, z, M, N, K);
    }
  }

  // ---- pool + head ----
  pool_mean<<<cdiv(B_ * D_, 256), 256, 0, stream>>>(z, y);
  head_mlp<<<B_, 256, 0, stream>>>(y, hn_s, hn_b, h1_w, h1_b, h2_w, h2_b, h3_w, h3_b, out);
}
